// AudioModelM1_74783970558075
// MI455X (gfx1250) — compile-verified, matrix operand order checked by reading, not run on hardware
//
#include <hip/hip_runtime.h>

// ---------------------------------------------------------------------------
// Mamba2 x2 + heads for MI455X (gfx1250, wave32, WMMA + Tensor Data Mover).
// GEMMs in bf16 WMMA (f32 accum), tiles staged by TDM (async, TENSORcnt);
// scan / norms / conv in fp32 VALU with TDM-double-buffered chunk staging.
// ---------------------------------------------------------------------------

typedef __attribute__((ext_vector_type(16))) __bf16 v16bf;
typedef __attribute__((ext_vector_type(8)))  float  v8f;
typedef __attribute__((ext_vector_type(4)))  unsigned int u32x4;
typedef __attribute__((ext_vector_type(8)))  int i32x8;
typedef __attribute__((ext_vector_type(4)))  int i32x4;

#ifndef __has_builtin
#define __has_builtin(x) 0
#endif
#if __has_builtin(__builtin_amdgcn_tensor_load_to_lds) && \
    __has_builtin(__builtin_amdgcn_s_wait_tensorcnt)
#define HAVE_TDM 1
#else
#define HAVE_TDM 0
#endif

#define D_MODEL   1024
#define D_STATE   64
#define D_CONV    4
#define D_INNER   2048
#define HEADDIM   64
#define NHEADS    32
#define D_CONVCH  (D_INNER + 2 * D_STATE)              // 2176
#define D_INPROJ  (2 * D_INNER + 2 * D_STATE + NHEADS) // 4256
#define NPROJ_PAD 4352                                 // 4256 padded to 128
#define BATCH     8
#define SEQ       2048
#define NTOK      (BATCH * SEQ)                        // 16384

// ---------------------------------------------------------------------------
// helpers
// ---------------------------------------------------------------------------
__device__ __forceinline__ float dev_sigmoid(float x) { return 1.0f / (1.0f + __expf(-x)); }

__device__ __forceinline__ v16bf lds_ld16(const __bf16* p) {
    union U { uint4 q[2]; v16bf v; } u;
    u.q[0] = *(const uint4*)(p);
    u.q[1] = *(const uint4*)(p + 8);
    return u.v;
}

__device__ __forceinline__ unsigned lds_off32(const void* p) {
    // LDS generic pointers: low 32 bits are the LDS byte offset
    return (unsigned)(uintptr_t)p;
}

#if HAVE_TDM
// Issue a 2-D TDM tile load: tile (tile_d1 rows x tile_d0 elems) from a
// row-major tensor (row stride stride0 elems), element size 2^ds_log2 bytes,
// into LDS at lds_off.  Optional LDS row padding (TDM pad fields).
__device__ __forceinline__ void tdm_load_2d(const void* gptr, unsigned lds_off,
                                            unsigned tensor_d0, unsigned tensor_d1,
                                            unsigned long long stride0,
                                            unsigned tile_d0, unsigned tile_d1,
                                            int ds_log2, int pad_en,
                                            int pad_interval_code, int pad_amount_code) {
    unsigned long long ga = (unsigned long long)(uintptr_t)gptr;
    u32x4 g0;
    g0.x = 1u;                                        // count=1, user descriptor
    g0.y = lds_off;                                   // lds_addr (bytes)
    g0.z = (unsigned)ga;                              // global_addr[31:0]
    g0.w = (unsigned)((ga >> 32) & 0x01FFFFFFu) | (2u << 30); // ga[56:32], type=2
    unsigned d0 = ((unsigned)ds_log2 << 16);          // wg_mask=0, data_size
    if (pad_en)
        d0 |= (1u << 20) | ((unsigned)pad_interval_code << 22)
            | ((unsigned)pad_amount_code << 25);
    i32x8 g1;
    g1[0] = (int)d0;
    g1[1] = (int)((tensor_d0 & 0xFFFFu) << 16);       // abar_addr=0 | tensor_dim0[15:0]
    g1[2] = (int)(((tensor_d0 >> 16) & 0xFFFFu) | ((tensor_d1 & 0xFFFFu) << 16));
    g1[3] = (int)(((tensor_d1 >> 16) & 0xFFFFu) | ((tile_d0 & 0xFFFFu) << 16));
    g1[4] = (int)(tile_d1 & 0xFFFFu);                 // tile_dim1 | tile_dim2=0
    g1[5] = (int)(unsigned)(stride0 & 0xFFFFFFFFull); // tensor_dim0_stride[31:0]
    g1[6] = (int)(unsigned)((stride0 >> 32) & 0xFFFFull); // stride[47:32] | dim1_stride lo=0
    g1[7] = 0;
    i32x4 z4 = (i32x4)0;
#if __clang_major__ >= 23
    i32x8 z8 = (i32x8)0;
    __builtin_amdgcn_tensor_load_to_lds(g0, g1, z4, z4, z8, 0);
#else
    __builtin_amdgcn_tensor_load_to_lds(g0, g1, z4, z4, 0);
#endif
}
#endif // HAVE_TDM

// ---------------------------------------------------------------------------
// fp32 -> bf16 conversion kernels
// ---------------------------------------------------------------------------
__global__ void __launch_bounds__(256) f32_to_bf16_kernel(const float* __restrict__ src,
                                                          __bf16* __restrict__ dst, size_t n) {
    size_t i = (size_t)blockIdx.x * 256 + threadIdx.x;
    if (i < n) dst[i] = (__bf16)src[i];
}

__global__ void __launch_bounds__(256) cvt_pad_inproj_kernel(const float* __restrict__ src,
                                                             __bf16* __restrict__ dst) {
    size_t i = (size_t)blockIdx.x * 256 + threadIdx.x;
    if (i < (size_t)NPROJ_PAD * D_MODEL) {
        size_t r = i >> 10; // /1024
        dst[i] = (r < D_INPROJ) ? (__bf16)src[i] : (__bf16)0.0f;
    }
}

// ---------------------------------------------------------------------------
// bf16 WMMA GEMM:  C[M,N] = A[M,K] * B[N,K]^T
// Block tile 128x128x64, 256 threads = 8 waves, wave tile 64x32 (4x2 wmma).
// TDM double-buffered LDS tiles: wave 0 issues 2-D descriptors; the hardware
// pad fields reproduce the 72-element LDS row stride (32 DW + 4 DW pad).
// ---------------------------------------------------------------------------
#define BLK_M 128
#define BLK_N 128
#define BLK_K 64
#define LDSK  72   // BLK_K + 8 pad (bf16 elems)

template <int EPI>
__global__ void __launch_bounds__(256)
gemm_bf16_wmma(const __bf16* __restrict__ A, const __bf16* __restrict__ B,
               float* __restrict__ Cf, __bf16* __restrict__ Cb,
               int M, int N, int K) {
    __shared__ __align__(16) __bf16 sA[2][BLK_M * LDSK];
    __shared__ __align__(16) __bf16 sB[2][BLK_N * LDSK];

    const int tid  = threadIdx.x;
    const int lane = tid & 31;
    const int wave = tid >> 5;      // 0..7
    const int wm   = wave >> 2;     // 0..1 -> M offset wm*64
    const int wn   = wave & 3;      // 0..3 -> N offset wn*32
    const int m_blk = blockIdx.y * BLK_M;
    const int n_blk = blockIdx.x * BLK_N;

    v8f acc[4][2];
#pragma unroll
    for (int mt = 0; mt < 4; ++mt)
#pragma unroll
        for (int nt = 0; nt < 2; ++nt) acc[mt][nt] = (v8f)(0.0f);

    const int krow  = lane & 15;
    const int khalf = (lane >> 4) * 16;

#if HAVE_TDM
    const bool issuer = (wave == 0);   // whole wave 0, EXEC all-ones
    // tile rows are 64 bf16 = 128 B = 16x 8B units -> pad_interval code 4;
    // pad 4 DWORDs (8 bf16)                        -> pad_amount  code 3.
    if (issuer) {
        tdm_load_2d(A + (size_t)m_blk * K, lds_off32(&sA[0][0]),
                    (unsigned)K, BLK_M, (unsigned long long)K, BLK_K, BLK_M, 1, 1, 4, 3);
        tdm_load_2d(B + (size_t)n_blk * K, lds_off32(&sB[0][0]),
                    (unsigned)K, BLK_N, (unsigned long long)K, BLK_K, BLK_N, 1, 1, 4, 3);
        __builtin_amdgcn_s_wait_tensorcnt(0);
    }
    __syncthreads();

    int cur = 0;
    for (int k0 = 0; k0 < K; k0 += BLK_K) {
        const int nxt = cur ^ 1;
        if (issuer && (k0 + BLK_K < K)) {
            tdm_load_2d(A + (size_t)m_blk * K + (k0 + BLK_K), lds_off32(&sA[nxt][0]),
                        (unsigned)K, BLK_M, (unsigned long long)K, BLK_K, BLK_M, 1, 1, 4, 3);
            tdm_load_2d(B + (size_t)n_blk * K + (k0 + BLK_K), lds_off32(&sB[nxt][0]),
                        (unsigned)K, BLK_N, (unsigned long long)K, BLK_K, BLK_N, 1, 1, 4, 3);
        }
#pragma unroll
        for (int kk = 0; kk < BLK_K; kk += 32) {
            v16bf afrag[4], bfrag[2];
#pragma unroll
            for (int mt = 0; mt < 4; ++mt)
                afrag[mt] = lds_ld16(sA[cur] + (wm * 64 + mt * 16 + krow) * LDSK + kk + khalf);
#pragma unroll
            for (int nt = 0; nt < 2; ++nt)
                bfrag[nt] = lds_ld16(sB[cur] + (wn * 32 + nt * 16 + krow) * LDSK + kk + khalf);
#pragma unroll
            for (int mt = 0; mt < 4; ++mt)
#pragma unroll
                for (int nt = 0; nt < 2; ++nt)
                    acc[mt][nt] = __builtin_amdgcn_wmma_f32_16x16x32_bf16(
                        false, afrag[mt], false, bfrag[nt],
                        (short)0, acc[mt][nt], false, false);
        }
        if (issuer) __builtin_amdgcn_s_wait_tensorcnt(0);
        __syncthreads();
        cur = nxt;
    }
#else  // !HAVE_TDM: synchronous staging fallback
    const int lrow = tid >> 3;
    const int lcol = (tid & 7) * 8;
    for (int k0 = 0; k0 < K; k0 += BLK_K) {
#pragma unroll
        for (int p = 0; p < 4; ++p) {
            int r = lrow + p * 32;
            uint4 va = *(const uint4*)(A + (size_t)(m_blk + r) * K + k0 + lcol);
            *(uint4*)(sA[0] + r * LDSK + lcol) = va;
            uint4 vb = *(const uint4*)(B + (size_t)(n_blk + r) * K + k0 + lcol);
            *(uint4*)(sB[0] + r * LDSK + lcol) = vb;
        }
        __syncthreads();
        if (k0 + BLK_K < K) {
            __builtin_prefetch(A + (size_t)(m_blk + lrow) * K + k0 + BLK_K + lcol, 0, 1);
            __builtin_prefetch(B + (size_t)(n_blk + lrow) * K + k0 + BLK_K + lcol, 0, 1);
        }
#pragma unroll
        for (int kk = 0; kk < BLK_K; kk += 32) {
            v16bf afrag[4], bfrag[2];
#pragma unroll
            for (int mt = 0; mt < 4; ++mt)
                afrag[mt] = lds_ld16(sA[0] + (wm * 64 + mt * 16 + krow) * LDSK + kk + khalf);
#pragma unroll
            for (int nt = 0; nt < 2; ++nt)
                bfrag[nt] = lds_ld16(sB[0] + (wn * 32 + nt * 16 + krow) * LDSK + kk + khalf);
#pragma unroll
            for (int mt = 0; mt < 4; ++mt)
#pragma unroll
                for (int nt = 0; nt < 2; ++nt)
                    acc[mt][nt] = __builtin_amdgcn_wmma_f32_16x16x32_bf16(
                        false, afrag[mt], false, bfrag[nt],
                        (short)0, acc[mt][nt], false, false);
        }
        __syncthreads();
    }
#endif

    // store: C/D layout — VGPR i holds M = i + (lane>=16 ? 8 : 0), N = lane&15
    const int rhalf = (lane >> 4) * 8;
    const int ccol  = n_blk + wn * 32 + (lane & 15);
#pragma unroll
    for (int mt = 0; mt < 4; ++mt)
#pragma unroll
        for (int nt = 0; nt < 2; ++nt)
#pragma unroll
            for (int i = 0; i < 8; ++i) {
                int r = m_blk + wm * 64 + mt * 16 + rhalf + i;
                int c = ccol + nt * 16;
                float v = acc[mt][nt][i];
                if (EPI == 0) {
                    Cf[(size_t)r * N + c] = v;
                } else {
                    const float SC = 1.0507009873554805f, AL = 1.6732632423543772f;
                    v = (v > 0.0f) ? SC * v : SC * AL * (__expf(v) - 1.0f);
                    Cb[(size_t)r * N + c] = (__bf16)v;
                }
            }
}

// ---------------------------------------------------------------------------
// depthwise causal conv (4 taps) + SiLU over xBC channels
// ---------------------------------------------------------------------------
__global__ void __launch_bounds__(256) conv_silu_kernel(const float* __restrict__ zx,
                                                        const float* __restrict__ cw,
                                                        const float* __restrict__ cb,
                                                        float* __restrict__ xconv) {
    size_t idx = (size_t)blockIdx.x * 256 + threadIdx.x;
    if (idx >= (size_t)NTOK * D_CONVCH) return;
    int c = (int)(idx % D_CONVCH);
    size_t bt = idx / D_CONVCH;
    int t = (int)(bt % SEQ);
    size_t b = bt / SEQ;
    float s = cb[c];
#pragma unroll
    for (int k = 0; k < D_CONV; ++k) {
        int tt = t + k - (D_CONV - 1);
        if (tt >= 0)
            s += zx[(b * SEQ + tt) * (size_t)NPROJ_PAD + D_INNER + c] * cw[c * D_CONV + k];
    }
    xconv[idx] = s * dev_sigmoid(s);   // silu
}

// ---------------------------------------------------------------------------
// dt = softplus(raw + dt_bias), dA = exp(dt * -exp(A_log));  [NTOK, 32]
// ---------------------------------------------------------------------------
__global__ void __launch_bounds__(256) dt_kernel(const float* __restrict__ zx,
                                                 const float* __restrict__ dt_bias,
                                                 const float* __restrict__ A_log,
                                                 float* __restrict__ dt,
                                                 float* __restrict__ dA) {
    size_t idx = (size_t)blockIdx.x * 256 + threadIdx.x;
    if (idx >= (size_t)NTOK * NHEADS) return;
    size_t row = idx >> 5;
    int hh = (int)(idx & 31);
    float raw = zx[row * NPROJ_PAD + (D_INPROJ - NHEADS) + hh] + dt_bias[hh];
    float d = (raw > 20.0f) ? raw : log1pf(__expf(raw));
    dt[idx] = d;
    dA[idx] = __expf(d * (-__expf(A_log[hh])));
}

// ---------------------------------------------------------------------------
// selective scan: one block per (batch, head); 256 threads keep 64x64 state.
// thread t owns (p = t>>2, n in [(t&3)*16, +16)). B/C/x chunk staging by TDM
// (three 16x64 fp32 tiles), double buffered against the 16-step FMA burst.
// ---------------------------------------------------------------------------
#define TCHUNK 16
__global__ void __launch_bounds__(256) scan_kernel(const float* __restrict__ xconv,
                                                   const float* __restrict__ dt,
                                                   const float* __restrict__ dA,
                                                   const float* __restrict__ Dp,
                                                   float* __restrict__ y) {
    const int b = blockIdx.x >> 5;
    const int h = blockIdx.x & 31;
    const int tid = threadIdx.x;
    const int p  = tid >> 2;
    const int nb = (tid & 3) << 4;

    __shared__ __align__(16) float sB[2][TCHUNK][D_STATE];
    __shared__ __align__(16) float sC[2][TCHUNK][D_STATE];
    __shared__ __align__(16) float sx[2][TCHUNK][HEADDIM];
    __shared__ float sdt[2][TCHUNK];
    __shared__ float sdA[2][TCHUNK];

    float hs[16];
#pragma unroll
    for (int i = 0; i < 16; ++i) hs[i] = 0.0f;

    const float Dv = Dp[h];
    const size_t base_bt = (size_t)b * SEQ;

#if HAVE_TDM
    const bool issuer = ((tid >> 5) == 0);
    // prologue: chunk 0 into buffer 0
    if (tid < TCHUNK) {
        size_t row = base_bt + tid;
        sdt[0][tid] = dt[row * NHEADS + h];
        sdA[0][tid] = dA[row * NHEADS + h];
    }
    if (issuer) {
        const float* r0 = xconv + base_bt * (size_t)D_CONVCH;
        tdm_load_2d(r0 + h * HEADDIM,          lds_off32(&sx[0][0][0]),
                    D_CONVCH, TCHUNK, D_CONVCH, HEADDIM, TCHUNK, 2, 0, 0, 0);
        tdm_load_2d(r0 + D_INNER,              lds_off32(&sB[0][0][0]),
                    D_CONVCH, TCHUNK, D_CONVCH, D_STATE, TCHUNK, 2, 0, 0, 0);
        tdm_load_2d(r0 + D_INNER + D_STATE,    lds_off32(&sC[0][0][0]),
                    D_CONVCH, TCHUNK, D_CONVCH, D_STATE, TCHUNK, 2, 0, 0, 0);
        __builtin_amdgcn_s_wait_tensorcnt(0);
    }
    __syncthreads();

    int cur = 0;
    for (int t0 = 0; t0 < SEQ; t0 += TCHUNK) {
        const int nxt = cur ^ 1;
        if (t0 + TCHUNK < SEQ) {
            if (tid < TCHUNK) {
                size_t row = base_bt + t0 + TCHUNK + tid;
                sdt[nxt][tid] = dt[row * NHEADS + h];
                sdA[nxt][tid] = dA[row * NHEADS + h];
            }
            if (issuer) {
                const float* r0 = xconv + (base_bt + t0 + TCHUNK) * (size_t)D_CONVCH;
                tdm_load_2d(r0 + h * HEADDIM,       lds_off32(&sx[nxt][0][0]),
                            D_CONVCH, TCHUNK, D_CONVCH, HEADDIM, TCHUNK, 2, 0, 0, 0);
                tdm_load_2d(r0 + D_INNER,           lds_off32(&sB[nxt][0][0]),
                            D_CONVCH, TCHUNK, D_CONVCH, D_STATE, TCHUNK, 2, 0, 0, 0);
                tdm_load_2d(r0 + D_INNER + D_STATE, lds_off32(&sC[nxt][0][0]),
                            D_CONVCH, TCHUNK, D_CONVCH, D_STATE, TCHUNK, 2, 0, 0, 0);
            }
        }
        for (int s = 0; s < TCHUNK; ++s) {
            float dAv = sdA[cur][s], dtv = sdt[cur][s];
            float xp = sx[cur][s][p];
            float coef = dtv * xp;
            float accv = 0.0f;
#pragma unroll
            for (int i = 0; i < 16; ++i) {
                hs[i] = fmaf(dAv, hs[i], coef * sB[cur][s][nb + i]);
                accv  = fmaf(hs[i], sC[cur][s][nb + i], accv);
            }
            accv += __shfl_xor(accv, 1, 32);
            accv += __shfl_xor(accv, 2, 32);
            if ((tid & 3) == 0) {
                size_t row = base_bt + t0 + s;
                y[row * (size_t)D_INNER + h * HEADDIM + p] = accv + Dv * xp;
            }
        }
        if (issuer) __builtin_amdgcn_s_wait_tensorcnt(0);
        __syncthreads();
        cur = nxt;
    }
#else  // !HAVE_TDM: synchronous staging fallback (single buffer)
    for (int t0 = 0; t0 < SEQ; t0 += TCHUNK) {
        for (int i = tid; i < TCHUNK * 64; i += 256) {
            int s = i >> 6, n = i & 63;
            const float* rp = xconv + (base_bt + t0 + s) * (size_t)D_CONVCH;
            sx[0][s][n] = rp[h * HEADDIM + n];
            sB[0][s][n] = rp[D_INNER + n];
            sC[0][s][n] = rp[D_INNER + D_STATE + n];
        }
        if (tid < TCHUNK) {
            size_t row = base_bt + t0 + tid;
            sdt[0][tid] = dt[row * NHEADS + h];
            sdA[0][tid] = dA[row * NHEADS + h];
        }
        __syncthreads();
        for (int s = 0; s < TCHUNK; ++s) {
            float dAv = sdA[0][s], dtv = sdt[0][s];
            float xp = sx[0][s][p];
            float coef = dtv * xp;
            float accv = 0.0f;
#pragma unroll
            for (int i = 0; i < 16; ++i) {
                hs[i] = fmaf(dAv, hs[i], coef * sB[0][s][nb + i]);
                accv  = fmaf(hs[i], sC[0][s][nb + i], accv);
            }
            accv += __shfl_xor(accv, 1, 32);
            accv += __shfl_xor(accv, 2, 32);
            if ((tid & 3) == 0) {
                size_t row = base_bt + t0 + s;
                y[row * (size_t)D_INNER + h * HEADDIM + p] = accv + Dv * xp;
            }
        }
        __syncthreads();
    }
#endif
}

// ---------------------------------------------------------------------------
// y = y * silu(z); RMSNorm over 2048; write bf16. one block per token.
// ---------------------------------------------------------------------------
__global__ void __launch_bounds__(256) gate_rmsnorm_kernel(const float* __restrict__ zx,
                                                           const float* __restrict__ y,
                                                           const float* __restrict__ norm_w,
                                                           __bf16* __restrict__ out) {
    const size_t row = blockIdx.x;
    __shared__ float vals[D_INNER];
    __shared__ float red[8];
    float ss = 0.0f;
    for (int c = threadIdx.x; c < D_INNER; c += 256) {
        float z = zx[row * NPROJ_PAD + c];
        float v = y[row * D_INNER + c] * (z * dev_sigmoid(z));
        vals[c] = v;
        ss += v * v;
    }
#pragma unroll
    for (int o = 16; o > 0; o >>= 1) ss += __shfl_xor(ss, o, 32);
    if ((threadIdx.x & 31) == 0) red[threadIdx.x >> 5] = ss;
    __syncthreads();
    if (threadIdx.x == 0) {
        float t = 0.0f;
#pragma unroll
        for (int i = 0; i < 8; ++i) t += red[i];
        red[0] = rsqrtf(t * (1.0f / D_INNER) + 1e-5f);
    }
    __syncthreads();
    float rs = red[0];
    for (int c = threadIdx.x; c < D_INNER; c += 256)
        out[row * D_INNER + c] = (__bf16)(vals[c] * rs * norm_w[c]);
}

// ---------------------------------------------------------------------------
// mean-pool over time
// ---------------------------------------------------------------------------
__global__ void __launch_bounds__(256) pool_kernel(const __bf16* __restrict__ act,
                                                   float* __restrict__ pooled) {
    int d = blockIdx.x * 256 + threadIdx.x;   // grid.x = 4
    int b = blockIdx.y;
    float s = 0.0f;
    for (int t = 0; t < SEQ; ++t)
        s += (float)act[((size_t)b * SEQ + t) * D_MODEL + d];
    pooled[b * D_MODEL + d] = s * (1.0f / SEQ);
}

// ---------------------------------------------------------------------------
// heads
// ---------------------------------------------------------------------------
__global__ void __launch_bounds__(32) heads_kernel(const float* __restrict__ pooled,
                                                   const float* __restrict__ w_emo,
                                                   const float* __restrict__ b_emo,
                                                   const float* __restrict__ w_sen,
                                                   const float* __restrict__ b_sen,
                                                   float* __restrict__ out) {
    int b = blockIdx.x;
    int j = threadIdx.x;
    if (j < 10) {
        const float* w = (j < 7) ? (w_emo + j * D_MODEL) : (w_sen + (j - 7) * D_MODEL);
        float s = (j < 7) ? b_emo[j] : b_sen[j - 7];
        const float* pv = pooled + b * D_MODEL;
        for (int i = 0; i < D_MODEL; ++i) s += pv[i] * w[i];
        out[b * 10 + j] = s;
    }
}

// ---------------------------------------------------------------------------
// launch
// ---------------------------------------------------------------------------
extern "C" void kernel_launch(void* const* d_in, const int* in_sizes, int n_in,
                              void* d_out, int out_size, void* d_ws, size_t ws_size,
                              hipStream_t stream) {
    (void)in_sizes; (void)n_in; (void)out_size; (void)ws_size;

    const float* x = (const float*)d_in[0];
    const float* w_emo = (const float*)d_in[17];
    const float* b_emo = (const float*)d_in[18];
    const float* w_sen = (const float*)d_in[19];
    const float* b_sen = (const float*)d_in[20];

    char* wsp = (char*)d_ws;
    auto carve = [&](size_t bytes) -> void* {
        void* p = (void*)wsp;
        wsp += (bytes + 255) & ~(size_t)255;
        return p;
    };
    __bf16* ACT   = (__bf16*)carve((size_t)NTOK * D_MODEL * 2);
    __bf16* YN    = (__bf16*)carve((size_t)NTOK * D_INNER * 2);
    float*  ZX    = (float*) carve((size_t)NTOK * NPROJ_PAD * 4);
    float*  XCONV = (float*) carve((size_t)NTOK * D_CONVCH * 4);
    float*  DT    = (float*) carve((size_t)NTOK * NHEADS * 4);
    float*  DA    = (float*) carve((size_t)NTOK * NHEADS * 4);
    float*  YBUF  = (float*) carve((size_t)NTOK * D_INNER * 4);
    __bf16* WI    = (__bf16*)carve((size_t)NPROJ_PAD * D_MODEL * 2);
    __bf16* WO    = (__bf16*)carve((size_t)D_MODEL * D_INNER * 2);
    float*  POOLED= (float*) carve((size_t)BATCH * D_MODEL * 4);

    {
        size_t n = (size_t)NTOK * D_MODEL;
        f32_to_bf16_kernel<<<dim3((unsigned)((n + 255) / 256)), 256, 0, stream>>>(x, ACT, n);
    }

    for (int l = 0; l < 2; ++l) {
        const float* in_proj_w = (const float*)d_in[1 + l * 8 + 0];
        const float* conv_w    = (const float*)d_in[1 + l * 8 + 1];
        const float* conv_b    = (const float*)d_in[1 + l * 8 + 2];
        const float* dt_bias   = (const float*)d_in[1 + l * 8 + 3];
        const float* A_log     = (const float*)d_in[1 + l * 8 + 4];
        const float* Dp        = (const float*)d_in[1 + l * 8 + 5];
        const float* norm_w    = (const float*)d_in[1 + l * 8 + 6];
        const float* out_projw = (const float*)d_in[1 + l * 8 + 7];

        {
            size_t n = (size_t)NPROJ_PAD * D_MODEL;
            cvt_pad_inproj_kernel<<<dim3((unsigned)((n + 255) / 256)), 256, 0, stream>>>(in_proj_w, WI);
            size_t m = (size_t)D_MODEL * D_INNER;
            f32_to_bf16_kernel<<<dim3((unsigned)((m + 255) / 256)), 256, 0, stream>>>(out_projw, WO, m);
        }

        gemm_bf16_wmma<0><<<dim3(NPROJ_PAD / BLK_N, NTOK / BLK_M), 256, 0, stream>>>(
            ACT, WI, ZX, nullptr, NTOK, NPROJ_PAD, D_MODEL);

        {
            size_t n = (size_t)NTOK * D_CONVCH;
            conv_silu_kernel<<<dim3((unsigned)((n + 255) / 256)), 256, 0, stream>>>(ZX, conv_w, conv_b, XCONV);
        }
        {
            size_t n = (size_t)NTOK * NHEADS;
            dt_kernel<<<dim3((unsigned)((n + 255) / 256)), 256, 0, stream>>>(ZX, dt_bias, A_log, DT, DA);
        }
        scan_kernel<<<dim3(BATCH * NHEADS), 256, 0, stream>>>(XCONV, DT, DA, Dp, YBUF);

        gate_rmsnorm_kernel<<<dim3(NTOK), 256, 0, stream>>>(ZX, YBUF, norm_w, YN);

        gemm_bf16_wmma<1><<<dim3(D_MODEL / BLK_N, NTOK / BLK_M), 256, 0, stream>>>(
            YN, WO, nullptr, ACT, NTOK, D_MODEL, D_INNER);
    }

    pool_kernel<<<dim3(D_MODEL / 256, BATCH), 256, 0, stream>>>(ACT, POOLED);
    heads_kernel<<<dim3(BATCH), 32, 0, stream>>>(POOLED, w_emo, b_emo, w_sen, b_sen, (float*)d_out);
}
